// PPOTransformerCommAgent_15582141350577
// MI455X (gfx1250) — matrix operand order repaired and checked by reading, not compiled, not run on hardware
//
#include <hip/hip_runtime.h>
#include <cstdint>
#include <cstddef>

// ---------------- model dimensions ----------------
#define D_MODEL 512
#define NHEAD   8
#define HDIM    64
#define NLAYER  4
#define MLPDIM  2048
#define BATCH   256
#define ROLL    64
#define ROWS    512        // 2 * BATCH (stacked [mem_tok, cur_tok])
#define ENC_ROWS 16384     // ROLL * BATCH
#define CHWDIM  768

typedef __bf16 bf16_t;
typedef bf16_t v16bf __attribute__((ext_vector_type(16)));
typedef float  v8f   __attribute__((ext_vector_type(8)));

union FragU {
  uint4 q[2];
  unsigned short us[16];
  v16bf v;
};

__device__ __forceinline__ unsigned short f2bf(float f) {
  unsigned int u = __float_as_uint(f);
  u += 0x7fffu + ((u >> 16) & 1u);   // round to nearest even
  return (unsigned short)(u >> 16);
}

// ---------------- conversion kernels ----------------
__global__ void k_cvt_scale(const float* __restrict__ src, unsigned short* __restrict__ dst,
                            float scale, unsigned int n) {
  unsigned int i = blockIdx.x * 256u + threadIdx.x;
  if (i < n) dst[i] = f2bf(src[i] * scale);
}

// dst[N, Kp] = transpose(src[K, N]) as bf16, zero-padded K -> Kp
__global__ void k_cvt_tr(const float* __restrict__ src, unsigned short* __restrict__ dst,
                         int K, int N, int Kp) {
  unsigned int i = blockIdx.x * 256u + threadIdx.x;
  unsigned int total = (unsigned int)N * (unsigned int)Kp;
  if (i >= total) return;
  int n = (int)(i / (unsigned int)Kp);
  int k = (int)(i % (unsigned int)Kp);
  dst[i] = (k < K) ? f2bf(src[(size_t)k * N + n]) : (unsigned short)0;
}

// ---------------- bf16 WMMA GEMM (weights pre-transposed) ----------------
// C[M,N] = epilogue(A[M,K] @ Wt[N,K]^T + bias) (+ resid)
// act: 0 = none, 1 = relu, 2 = gelu(exact)
// M multiple of 128; K multiple of 32. N guarded (zero-filled B rows past N).
__global__ __launch_bounds__(256)
void k_gemm_bf16(const unsigned short* __restrict__ A,   // [M,K] row-major
                 const unsigned short* __restrict__ Wt,  // [N,K] row-major (W transposed)
                 const float* __restrict__ bias,
                 const float* __restrict__ resid,
                 float* __restrict__ outF,
                 unsigned short* __restrict__ outH, int ldH,
                 int M, int N, int K, int act) {
  __shared__ unsigned short As[2][128 * 32]; // [m][k]
  __shared__ unsigned short Bs[2][64 * 32];  // [n][k]  (column of W contiguous in k)
  const int tid  = threadIdx.x;
  const int wv   = tid >> 5;
  const int lane = tid & 31;
  const int m0 = blockIdx.y * 128;
  const int n0 = blockIdx.x * 64;
  const int waveM = (wv >> 1) * 32;
  const int waveN = (wv & 1) * 32;
  const int hi  = (lane >= 16) ? 1 : 0;
  const int l16 = lane & 15;
  (void)M;

  auto stageA = [&](int k0, int buf) {
    for (int it = 0; it < 2; ++it) {
      int v   = tid + it * 256;
      int row = v >> 2;
      int kv  = (v & 3) * 8;
      *(uint4*)(&As[buf][row * 32 + kv]) =
          *(const uint4*)(&A[(size_t)(m0 + row) * K + k0 + kv]);
    }
  };
  auto stageB = [&](int k0, int buf) {
    int n  = tid >> 2;            // 0..63
    int kv = (tid & 3) * 8;       // 0,8,16,24
    int gn = n0 + n;
    uint4 val;
    val.x = val.y = val.z = val.w = 0u;
    if (gn < N) val = *(const uint4*)(&Wt[(size_t)gn * K + k0 + kv]);
    *(uint4*)(&Bs[buf][n * 32 + kv]) = val;
  };

  v8f acc[2][2];
  for (int i = 0; i < 2; ++i)
    for (int j = 0; j < 2; ++j)
      for (int r = 0; r < 8; ++r) acc[i][j][r] = 0.0f;

  stageA(0, 0);
  stageB(0, 0);
  __syncthreads();

  int p = 0;
  for (int k0 = 0; k0 < K; k0 += 32) {
    const int nk = k0 + 32;
    if (nk < K) {                      // stage next tile into the other buffer
      stageA(nk, p ^ 1);
      stageB(nk, p ^ 1);
    }
    if (k0 + 64 < K) {                 // prefetch two tiles ahead (global_prefetch_b8)
      __builtin_prefetch(&A[(size_t)(m0 + (tid >> 2)) * K + k0 + 64 + (tid & 3) * 8], 0, 1);
      __builtin_prefetch(&Wt[(size_t)(n0 + (tid >> 2)) * K + k0 + 64 + (tid & 3) * 8], 0, 1);
    }

    // fragments per CDNA5 16-bit WMMA layouts (all b128 LDS loads)
    FragU fa[2], fb[2];
    for (int i = 0; i < 2; ++i) {
      int mrow = waveM + i * 16 + l16;
      int base = mrow * 32 + (hi ? 8 : 0);  // lane<16: K 0..7/16..23 ; lane>=16: K 8..15/24..31
      fa[i].q[0] = *(const uint4*)(&As[p][base]);
      fa[i].q[1] = *(const uint4*)(&As[p][base + 16]);
    }
    for (int j = 0; j < 2; ++j) {
      int col  = waveN + j * 16 + l16;
      int base = col * 32 + (hi ? 8 : 0);   // lane<16: K 0..7/16..23 ; lane>=16: K 8..15/24..31
      fb[j].q[0] = *(const uint4*)(&Bs[p][base]);
      fb[j].q[1] = *(const uint4*)(&Bs[p][base + 16]);
    }
    for (int i = 0; i < 2; ++i)
      for (int j = 0; j < 2; ++j)
        acc[i][j] = __builtin_amdgcn_wmma_f32_16x16x32_bf16(
            false, fa[i].v, false, fb[j].v, (short)0, acc[i][j], false, false);
    __syncthreads();                   // single barrier per K-step (ping-pong buffers)
    p ^= 1;
  }

  // epilogue: C layout => VGPR r: lane<16 -> M=r, lane>=16 -> M=8+r ; N = lane&15
  for (int i = 0; i < 2; ++i) {
    for (int j = 0; j < 2; ++j) {
      int n = n0 + waveN + j * 16 + l16;
      if (n >= N) continue;
      float bval = bias ? bias[n] : 0.0f;
      for (int r = 0; r < 8; ++r) {
        int m = m0 + waveM + i * 16 + r + (hi ? 8 : 0);
        float val = acc[i][j][r] + bval;
        if (act == 1)      val = fmaxf(val, 0.0f);
        else if (act == 2) val = 0.5f * val * (1.0f + erff(val * 0.70710678118f));
        if (resid) val += resid[(size_t)m * N + n];
        if (outF)  outF[(size_t)m * N + n] = val;
        if (outH)  outH[(size_t)m * ldH + n] = f2bf(val);
      }
    }
  }
}

// ---------------- LayerNorm (f32 in -> bf16 out), one row of D=512 per block ----------------
__global__ __launch_bounds__(256)
void k_ln(const float* __restrict__ x, const float* __restrict__ g,
          const float* __restrict__ b, unsigned short* __restrict__ h) {
  const int row = blockIdx.x;
  const int tid = threadIdx.x;
  __shared__ float s1[256], s2[256];
  float v0 = x[(size_t)row * D_MODEL + tid];
  float v1 = x[(size_t)row * D_MODEL + 256 + tid];
  s1[tid] = v0 + v1;
  s2[tid] = v0 * v0 + v1 * v1;
  __syncthreads();
  for (int off = 128; off; off >>= 1) {
    if (tid < off) { s1[tid] += s1[tid + off]; s2[tid] += s2[tid + off]; }
    __syncthreads();
  }
  float mean = s1[0] * (1.0f / D_MODEL);
  float var  = s2[0] * (1.0f / D_MODEL) - mean * mean;
  float rs   = rsqrtf(var + 1e-5f);
  h[(size_t)row * D_MODEL + tid]       = f2bf((v0 - mean) * rs * g[tid] + b[tid]);
  h[(size_t)row * D_MODEL + 256 + tid] = f2bf((v1 - mean) * rs * g[tid + 256] + b[tid + 256]);
}

// ---------------- 2x2 attention per (batch, head) ----------------
__global__ __launch_bounds__(64)
void k_attn(const float* __restrict__ qkv, unsigned short* __restrict__ abuf) {
  const int p = blockIdx.x;          // 0 .. B*NH-1
  const int b = p & (BATCH - 1);
  const int h = p >> 8;
  const int d = threadIdx.x;         // 0..63
  const float* r0 = qkv + (size_t)b * (3 * D_MODEL) + h * HDIM;           // s=0 row
  const float* r1 = qkv + (size_t)(BATCH + b) * (3 * D_MODEL) + h * HDIM; // s=1 row
  float q0 = r0[d], q1 = r1[d];
  float k0 = r0[D_MODEL + d], k1 = r1[D_MODEL + d];
  float v0 = r0[2 * D_MODEL + d], v1 = r1[2 * D_MODEL + d];
  __shared__ float red[4][64];
  red[0][d] = q0 * k0; red[1][d] = q0 * k1; red[2][d] = q1 * k0; red[3][d] = q1 * k1;
  __syncthreads();
  for (int off = 32; off; off >>= 1) {
    if (d < off)
      for (int j = 0; j < 4; ++j) red[j][d] += red[j][d + off];
    __syncthreads();
  }
  const float sc = 0.125f;           // 1/sqrt(64)
  float s00 = red[0][0] * sc, s01 = red[1][0] * sc;
  float s10 = red[2][0] * sc, s11 = red[3][0] * sc;
  float m0r = fmaxf(s00, s01), m1r = fmaxf(s10, s11);
  float e00 = __expf(s00 - m0r), e01 = __expf(s01 - m0r);
  float e10 = __expf(s10 - m1r), e11 = __expf(s11 - m1r);
  float i0 = 1.0f / (e00 + e01), i1 = 1.0f / (e10 + e11);
  float a0 = (e00 * i0) * v0 + (e01 * i0) * v1;
  float a1 = (e10 * i1) * v0 + (e11 * i1) * v1;
  abuf[(size_t)b * D_MODEL + h * HDIM + d]           = f2bf(a0);
  abuf[(size_t)(BATCH + b) * D_MODEL + h * HDIM + d] = f2bf(a1);
}

// ---------------- loc/msg encoders -> concat buffer cols 16..63 ----------------
__global__ void k_locmsg(const float* __restrict__ location, const int* __restrict__ message,
                         const float* __restrict__ emb, const float* __restrict__ Wmsg,
                         const float* __restrict__ bmsg, const float* __restrict__ Wloc,
                         const float* __restrict__ bloc, unsigned short* __restrict__ cat) {
  int row = blockIdx.x * 256 + threadIdx.x;
  if (row >= ENC_ROWS) return;
  float lx = location[(size_t)row * 2] * (1.0f / 32.0f);
  float ly = location[(size_t)row * 2 + 1] * (1.0f / 32.0f);
  unsigned short* c = cat + (size_t)row * 64;
  for (int j = 0; j < 4; ++j)
    c[16 + j] = f2bf(lx * Wloc[j] + ly * Wloc[4 + j] + bloc[j]);
  const float* e = emb + (size_t)message[row] * 32;
  for (int j = 0; j < 32; ++j) {
    float s = bmsg[j];
    for (int i = 0; i < 32; ++i) s += e[i] * Wmsg[i * 32 + j];
    c[20 + j] = f2bf(fmaxf(s, 0.0f));
  }
  for (int j = 52; j < 64; ++j) c[j] = 0;
}

// ---------------- per-step epilogue: write outs[t], new_mem = (1-done)*out ----------------
__global__ void k_update_mem(const float* __restrict__ x, const int* __restrict__ done_t,
                             float* __restrict__ out_t, float* __restrict__ memf,
                             unsigned short* __restrict__ membf) {
  int idx = blockIdx.x * 256 + threadIdx.x;   // < BATCH * D_MODEL
  int b = idx >> 9;
  int dv = idx & (D_MODEL - 1);
  float o = x[(size_t)(BATCH + b) * D_MODEL + dv];
  out_t[(size_t)b * D_MODEL + dv] = o;
  float m = (1.0f - (float)done_t[b]) * o;
  memf[idx]  = m;
  membf[idx] = f2bf(m);
}

// ---------------- host ----------------
extern "C" void kernel_launch(void* const* d_in, const int* in_sizes, int n_in,
                              void* d_out, int out_size, void* d_ws, size_t ws_size,
                              hipStream_t stream) {
  (void)in_sizes; (void)n_in; (void)out_size; (void)ws_size;
  const float* image      = (const float*)d_in[0];
  const float* location   = (const float*)d_in[1];
  const int*   message    = (const int*)  d_in[2];
  const int*   done       = (const int*)  d_in[3];
  const float* memory_in  = (const float*)d_in[4];
  const float* W_v1 = (const float*)d_in[5];  const float* b_v1 = (const float*)d_in[6];
  const float* W_v2 = (const float*)d_in[7];  const float* b_v2 = (const float*)d_in[8];
  const float* W_v3 = (const float*)d_in[9];  const float* b_v3 = (const float*)d_in[10];
  const float* W_v4 = (const float*)d_in[11]; const float* b_v4 = (const float*)d_in[12];
  const float* emb_table = (const float*)d_in[13];
  const float* W_msg = (const float*)d_in[14]; const float* b_msg = (const float*)d_in[15];
  const float* W_loc = (const float*)d_in[16]; const float* b_loc = (const float*)d_in[17];
  const float* W_in  = (const float*)d_in[18]; const float* b_in  = (const float*)d_in[19];
  const float* W_mem = (const float*)d_in[20];
  const float* ln1_g = (const float*)d_in[21]; const float* ln1_b = (const float*)d_in[22];
  const float* attn_in_w  = (const float*)d_in[23]; const float* attn_in_b  = (const float*)d_in[24];
  const float* attn_out_w = (const float*)d_in[25]; const float* attn_out_b = (const float*)d_in[26];
  const float* ln2_g = (const float*)d_in[27]; const float* ln2_b = (const float*)d_in[28];
  const float* mlp_w1 = (const float*)d_in[29]; const float* mlp_b1 = (const float*)d_in[30];
  const float* mlp_w2 = (const float*)d_in[31]; const float* mlp_b2 = (const float*)d_in[32];
  float* out = (float*)d_out;

  // --- workspace arena ---
  char* base = (char*)d_ws;
  size_t off = 0;
  auto allocb = [&](size_t bytes) -> void* {
    off = (off + 255) & ~(size_t)255;
    void* p = base + off;
    off += bytes;
    return p;
  };
  auto allocH = [&](size_t n) { return (unsigned short*)allocb(n * 2); };
  auto allocF = [&](size_t n) { return (float*)allocb(n * 4); };

  // transposed bf16 weights [N, K]
  unsigned short* wv1b  = allocH((size_t)256 * CHWDIM);
  unsigned short* wv2b  = allocH(256 * 256);
  unsigned short* wv3b  = allocH(128 * 256);
  unsigned short* wv4b  = allocH(16 * 128);
  unsigned short* winb  = allocH((size_t)D_MODEL * 64);            // K padded 52 -> 64
  unsigned short* wmemb = allocH((size_t)D_MODEL * D_MODEL);
  unsigned short* wqkvb = allocH((size_t)NLAYER * 3 * D_MODEL * D_MODEL);
  unsigned short* woutb = allocH((size_t)NLAYER * D_MODEL * D_MODEL);
  unsigned short* w1b   = allocH((size_t)NLAYER * MLPDIM * D_MODEL);
  unsigned short* w2b   = allocH((size_t)NLAYER * D_MODEL * MLPDIM);
  unsigned short* imgin = allocH((size_t)ENC_ROWS * CHWDIM);
  unsigned short* e1    = allocH((size_t)ENC_ROWS * 256);
  unsigned short* e2    = allocH((size_t)ENC_ROWS * 256);
  unsigned short* e3    = allocH((size_t)ENC_ROWS * 128);
  unsigned short* cat   = allocH((size_t)ENC_ROWS * 64);
  float*          tok   = allocF((size_t)ENC_ROWS * D_MODEL);
  float*          x     = allocF((size_t)ROWS * D_MODEL);
  unsigned short* h     = allocH((size_t)ROWS * D_MODEL);
  float*          qkv   = allocF((size_t)ROWS * 3 * D_MODEL);
  unsigned short* abuf  = allocH((size_t)ROWS * D_MODEL);
  unsigned short* ubuf  = allocH((size_t)ROWS * MLPDIM);
  unsigned short* membf = allocH((size_t)BATCH * D_MODEL);
  float*          memf  = allocF((size_t)BATCH * D_MODEL);

  auto cvt = [&](const float* s, unsigned short* d, float sc, unsigned int n) {
    k_cvt_scale<<<dim3((n + 255u) / 256u), dim3(256), 0, stream>>>(s, d, sc, n);
  };
  auto cvt_tr = [&](const float* s, unsigned short* d, int K, int N, int Kp) {
    unsigned int tot = (unsigned int)N * (unsigned int)Kp;
    k_cvt_tr<<<dim3((tot + 255u) / 256u), dim3(256), 0, stream>>>(s, d, K, N, Kp);
  };
  auto gemm = [&](const unsigned short* A, const unsigned short* Wt, const float* bias,
                  const float* resid, float* outF, unsigned short* outH, int ldH,
                  int M, int N, int K, int act) {
    dim3 g((unsigned)((N + 63) / 64), (unsigned)(M / 128));
    k_gemm_bf16<<<g, dim3(256), 0, stream>>>(A, Wt, bias, resid, outF, outH, ldH, M, N, K, act);
  };

  // --- weight conversion to transposed bf16 (once per launch) ---
  cvt_tr(W_v1, wv1b, CHWDIM, 256, CHWDIM);
  cvt_tr(W_v2, wv2b, 256, 256, 256);
  cvt_tr(W_v3, wv3b, 256, 128, 256);
  cvt_tr(W_v4, wv4b, 128, 16, 128);
  cvt_tr(W_in, winb, 52, D_MODEL, 64);
  cvt_tr(W_mem, wmemb, D_MODEL, D_MODEL, D_MODEL);
  for (int l = 0; l < NLAYER; ++l) {
    cvt_tr(attn_in_w  + (size_t)l * D_MODEL * 3 * D_MODEL,
           wqkvb + (size_t)l * 3 * D_MODEL * D_MODEL, D_MODEL, 3 * D_MODEL, D_MODEL);
    cvt_tr(attn_out_w + (size_t)l * D_MODEL * D_MODEL,
           woutb + (size_t)l * D_MODEL * D_MODEL, D_MODEL, D_MODEL, D_MODEL);
    cvt_tr(mlp_w1 + (size_t)l * D_MODEL * MLPDIM,
           w1b + (size_t)l * MLPDIM * D_MODEL, D_MODEL, MLPDIM, D_MODEL);
    cvt_tr(mlp_w2 + (size_t)l * MLPDIM * D_MODEL,
           w2b + (size_t)l * D_MODEL * MLPDIM, MLPDIM, D_MODEL, MLPDIM);
  }

  // --- encoder (parallel over all L*B tokens) ---
  cvt(image, imgin, 1.0f / 255.0f, (unsigned int)((size_t)ENC_ROWS * CHWDIM));
  gemm(imgin, wv1b, b_v1, nullptr, nullptr, e1, 256, ENC_ROWS, 256, CHWDIM, 1);
  gemm(e1, wv2b, b_v2, nullptr, nullptr, e2, 256, ENC_ROWS, 256, 256, 1);
  gemm(e2, wv3b, b_v3, nullptr, nullptr, e3, 128, ENC_ROWS, 128, 256, 1);
  gemm(e3, wv4b, b_v4, nullptr, nullptr, cat, 64, ENC_ROWS, 16, 128, 1);   // cols 0..15 of cat
  k_locmsg<<<dim3(ENC_ROWS / 256), dim3(256), 0, stream>>>(
      location, message, emb_table, W_msg, b_msg, W_loc, b_loc, cat);      // cols 16..63
  gemm(cat, winb, b_in, nullptr, tok, nullptr, 0, ENC_ROWS, D_MODEL, 64, 0);

  // --- recurrent scan over L steps ---
  cvt(memory_in, membf, 1.0f, BATCH * D_MODEL);
  for (int t = 0; t < ROLL; ++t) {
    // x[0:256]   = mem @ W_mem
    gemm(membf, wmemb, nullptr, nullptr, x, nullptr, 0, BATCH, D_MODEL, D_MODEL, 0);
    // x[256:512] = tok[t]
    hipMemcpyAsync(x + (size_t)BATCH * D_MODEL, tok + (size_t)t * BATCH * D_MODEL,
                   (size_t)BATCH * D_MODEL * sizeof(float), hipMemcpyDeviceToDevice, stream);
    for (int l = 0; l < NLAYER; ++l) {
      k_ln<<<dim3(ROWS), dim3(256), 0, stream>>>(x, ln1_g + l * D_MODEL, ln1_b + l * D_MODEL, h);
      gemm(h, wqkvb + (size_t)l * 3 * D_MODEL * D_MODEL, attn_in_b + l * 3 * D_MODEL,
           nullptr, qkv, nullptr, 0, ROWS, 3 * D_MODEL, D_MODEL, 0);
      k_attn<<<dim3(BATCH * NHEAD), dim3(64), 0, stream>>>(qkv, abuf);
      gemm(abuf, woutb + (size_t)l * D_MODEL * D_MODEL, attn_out_b + l * D_MODEL,
           x, x, nullptr, 0, ROWS, D_MODEL, D_MODEL, 0);                   // x += attn_out
      k_ln<<<dim3(ROWS), dim3(256), 0, stream>>>(x, ln2_g + l * D_MODEL, ln2_b + l * D_MODEL, h);
      gemm(h, w1b + (size_t)l * MLPDIM * D_MODEL, mlp_b1 + l * MLPDIM,
           nullptr, nullptr, ubuf, MLPDIM, ROWS, MLPDIM, D_MODEL, 2);      // gelu
      gemm(ubuf, w2b + (size_t)l * D_MODEL * MLPDIM, mlp_b2 + l * D_MODEL,
           x, x, nullptr, 0, ROWS, D_MODEL, MLPDIM, 0);                    // x += mlp
    }
    k_update_mem<<<dim3((BATCH * D_MODEL) / 256), dim3(256), 0, stream>>>(
        x, done + (size_t)t * BATCH, out + (size_t)t * BATCH * D_MODEL, memf, membf);
  }
  // final memory appended after hidden_flat
  hipMemcpyAsync(out + (size_t)ROLL * BATCH * D_MODEL, memf,
                 (size_t)BATCH * D_MODEL * sizeof(float), hipMemcpyDeviceToDevice, stream);
}
